// ResLiteMLA_14929306321344
// MI455X (gfx1250) — matrix operand order repaired and checked by reading, not compiled, not run on hardware
//
#include <hip/hip_runtime.h>

// ---------------------------------------------------------------------------
// ResLiteMLA for MI455X (gfx1250, wave32, WGP).
//  - qkv GEMM  : W_qkv(768x256)  @ X(256x4096)   per batch  (WMMA f16->f32)
//  - proj GEMM : W_proj(256x512) @ MLA(512x4096) per batch  (WMMA f16->f32)
// GEMM scheme: block = 128 threads (4 waves). Block tile 64M x 64N.
//   * B panel (32K x 64N) staged in LDS per K-step, transposed (row = n,
//     40-half padded stride -> conflict-free ds_load_b128 fragment reads).
//   * A pre-converted to f16 in d_ws; fragments are aligned b128 loads.
//   * Each wave holds 4 f32 accumulators (16M x 64N) -> 4 WMMAs per K-step
//     off one A fragment.
// ---------------------------------------------------------------------------

typedef __attribute__((ext_vector_type(16))) _Float16 v16h;
typedef __attribute__((ext_vector_type(8)))  _Float16 v8h;
typedef __attribute__((ext_vector_type(8)))  float    v8f;

#define BATCH 16
#define CIN   256
#define HH    64
#define WW    64
#define HW    4096          // H*W
#define TD3   768           // 3*TD
#define CMS   1536          // qkv_ms channels = qkv(768) + agg(768)
#define NHEAD 64
#define DIMQ  8
#define COUT_MLA 512

#define BROW  40            // padded LDS row stride (halfs) for the B panel

// ---------------------------------------------------------------------------
// Shared GEMM core: all 128 threads cooperate.  Computes the wave's
// 16M x 64N accumulator set.  sB is the 64x(BROW) half LDS panel.
// ---------------------------------------------------------------------------
__device__ __forceinline__ void gemm_tiles(const _Float16* __restrict__ A16, int lda,
                                           const float* __restrict__ Bm, int ldb,
                                           int tileM, int tileN, int K,
                                           _Float16* sB, v8f acc[4])
{
    const int lane = threadIdx.x & 31;
    const int half = lane >> 4;
    const int l16  = lane & 15;

    for (int kb = 0; kb < K; kb += 32) {
        // ---- stage B panel [kb..kb+32) x [tileN..tileN+64) into LDS (f16,
        //      transposed: sB[n*BROW + k]).  Coalesced global reads.
        #pragma unroll
        for (int i = 0; i < 16; ++i) {
            const int flat = i * 128 + threadIdx.x;   // 0..2047
            const int k = flat >> 6;                  // 0..31
            const int n = flat & 63;                  // 0..63
            sB[n * BROW + k] =
                (_Float16)Bm[(size_t)(kb + k) * ldb + tileN + n];
        }
        __syncthreads();

        // ---- A fragment: K = kb + half*8 + {0..7, 16..23}; aligned b128 x2
        const _Float16* arow = A16 + (size_t)(tileM + l16) * lda + kb + half * 8;
        const v8h a_lo = *(const v8h*)(arow);
        const v8h a_hi = *(const v8h*)(arow + 16);
        const v16h av = __builtin_shufflevector(a_lo, a_hi,
            0, 1, 2, 3, 4, 5, 6, 7, 8, 9, 10, 11, 12, 13, 14, 15);

        // ---- 4 WMMAs reuse the A fragment across the 64-wide N panel
        #pragma unroll
        for (int nt = 0; nt < 4; ++nt) {
            const _Float16* brow = sB + (size_t)(nt * 16 + l16) * BROW + half * 8;
            const v8h b_lo = *(const v8h*)(brow);       // K = half*8 + 0..7
            const v8h b_hi = *(const v8h*)(brow + 16);  // K = half*8 + 16..23
            const v16h bv = __builtin_shufflevector(b_lo, b_hi,
                0, 1, 2, 3, 4, 5, 6, 7, 8, 9, 10, 11, 12, 13, 14, 15);
            acc[nt] = __builtin_amdgcn_wmma_f32_16x16x32_f16(
                false, av, false, bv, (short)0, acc[nt], false, false);
        }
        __syncthreads();
    }
}

// ---------------------- weight f32 -> f16 pre-conversion --------------------
__global__ void k_cvt_f16(const float* __restrict__ src,
                          _Float16* __restrict__ dst, int n)
{
    const int i = blockIdx.x * 256 + threadIdx.x;
    if (i < n) dst[i] = (_Float16)src[i];
}

// ------------------------- Stage 1: qkv = W_qkv @ X -------------------------
// grid: (HW/64, TD3/64, BATCH), block 128
__global__ void k_qkv_gemm(const float* __restrict__ x,
                           const _Float16* __restrict__ wq16,
                           float* __restrict__ qkv_ms)
{
    __shared__ _Float16 sB[64 * BROW];
    const int wave  = threadIdx.x >> 5;
    const int lane  = threadIdx.x & 31;
    const int tileN = blockIdx.x * 64;
    const int tileM = blockIdx.y * 64 + wave * 16;
    const int b     = blockIdx.z;

    const float* Bm = x + (size_t)b * CIN * HW;
    float*       Cc = qkv_ms + (size_t)b * CMS * HW;

    v8f acc[4];
    #pragma unroll
    for (int nt = 0; nt < 4; ++nt)
        acc[nt] = (v8f){0.f, 0.f, 0.f, 0.f, 0.f, 0.f, 0.f, 0.f};

    gemm_tiles(wq16, CIN, Bm, HW, tileM, tileN, CIN, sB, acc);

    const int half = lane >> 4, l16 = lane & 15;
    #pragma unroll
    for (int nt = 0; nt < 4; ++nt) {
        const int n = tileN + nt * 16 + l16;
        #pragma unroll
        for (int r = 0; r < 8; ++r)
            Cc[(size_t)(tileM + half * 8 + r) * HW + n] = acc[nt][r];
    }
}

// ---------------- Stage 2: depthwise 5x5 conv + 8x8 group mix ---------------
__global__ void k_dw_mix(const float* __restrict__ qkv_ms_r,
                         const float* __restrict__ w_dw,   // (768,1,5,5)
                         const float* __restrict__ w_pw,   // (768,8)
                         float* __restrict__ qkv_ms_w)
{
    const int idx = blockIdx.x * blockDim.x + threadIdx.x;  // B*96*HW threads
    const int p = idx & (HW - 1);
    const int g = (idx >> 12) % 96;
    const int b = idx / (96 * HW);
    const int h = p >> 6, w = p & 63;

    const float* src = qkv_ms_r + (size_t)b * CMS * HW;

    float dwv[8];
#pragma unroll
    for (int i = 0; i < 8; ++i) {
        const int c = g * 8 + i;
        const float* ch = src + (size_t)c * HW;
        const float* wt = w_dw + c * 25;
        float s = 0.f;
#pragma unroll
        for (int dy = 0; dy < 5; ++dy) {
            const int hh = h + dy - 2;
            if (hh < 0 || hh >= HH) continue;
#pragma unroll
            for (int dx = 0; dx < 5; ++dx) {
                const int wx = w + dx - 2;
                if (wx < 0 || wx >= WW) continue;
                s += ch[hh * WW + wx] * wt[dy * 5 + dx];
            }
        }
        dwv[i] = s;
    }

    float* dst = qkv_ms_w + (size_t)b * CMS * HW + (size_t)(TD3 + g * 8) * HW + p;
#pragma unroll
    for (int o = 0; o < 8; ++o) {
        float s = 0.f;
#pragma unroll
        for (int i = 0; i < 8; ++i)
            s += w_pw[(g * 8 + o) * 8 + i] * dwv[i];
        dst[(size_t)o * HW] = s;
    }
}

// ----------------------- Stage 3: lite MLA per (b, head) --------------------
__global__ void k_lite_mla(const float* __restrict__ qkv_ms,
                           float* __restrict__ mla)
{
    __shared__ float s_vk[72];
    const int hd = blockIdx.x;
    const int b  = blockIdx.y;
    const int t  = threadIdx.x;

    const float* base = qkv_ms + (size_t)b * CMS * HW + (size_t)hd * 24 * HW;
    const float* qp = base;
    const float* kp = base + 8 * HW;
    const float* vp = base + 16 * HW;

    if (t < 72) s_vk[t] = 0.f;
    __syncthreads();

    float vk[9][8];
#pragma unroll
    for (int e = 0; e < 9; ++e)
#pragma unroll
        for (int d = 0; d < 8; ++d) vk[e][d] = 0.f;

    for (int p = t; p < HW; p += 256) {
        float kv[8], vv[8];
#pragma unroll
        for (int d = 0; d < 8; ++d) {
            const float kk = kp[(size_t)d * HW + p];
            kv[d] = kk > 0.f ? kk : 0.f;
            vv[d] = vp[(size_t)d * HW + p];
        }
#pragma unroll
        for (int e = 0; e < 8; ++e)
#pragma unroll
            for (int d = 0; d < 8; ++d)
                vk[e][d] += vv[e] * kv[d];
#pragma unroll
        for (int d = 0; d < 8; ++d)
            vk[8][d] += kv[d];                      // ones-row of v
    }

#pragma unroll
    for (int e = 0; e < 9; ++e)
#pragma unroll
        for (int d = 0; d < 8; ++d)
            atomicAdd(&s_vk[e * 8 + d], vk[e][d]);  // ds_add_f32
    __syncthreads();

    float fvk[9][8];
#pragma unroll
    for (int e = 0; e < 9; ++e)
#pragma unroll
        for (int d = 0; d < 8; ++d) fvk[e][d] = s_vk[e * 8 + d];

    float* dst = mla + (size_t)b * COUT_MLA * HW + (size_t)hd * DIMQ * HW;
    for (int p = t; p < HW; p += 256) {
        float qv[8];
#pragma unroll
        for (int d = 0; d < 8; ++d) {
            const float qq = qp[(size_t)d * HW + p];
            qv[d] = qq > 0.f ? qq : 0.f;
        }
        float den = 1e-15f;
#pragma unroll
        for (int d = 0; d < 8; ++d) den += fvk[8][d] * qv[d];
        const float inv = 1.f / den;
#pragma unroll
        for (int e = 0; e < 8; ++e) {
            float s = 0.f;
#pragma unroll
            for (int d = 0; d < 8; ++d) s += fvk[e][d] * qv[d];
            dst[(size_t)e * HW + p] = s * inv;
        }
    }
}

// ------------- Stage 4: proj GEMM + BatchNorm affine + residual -------------
// grid: (HW/64, CIN/64, BATCH), block 128
__global__ void k_proj_bn_res(const float* __restrict__ mla,
                              const _Float16* __restrict__ wp16,  // (256,512) f16
                              const float* __restrict__ gamma,
                              const float* __restrict__ beta,
                              const float* __restrict__ mean,
                              const float* __restrict__ var,
                              const float* __restrict__ x,
                              float* __restrict__ out)
{
    __shared__ _Float16 sB[64 * BROW];
    const int wave  = threadIdx.x >> 5;
    const int lane  = threadIdx.x & 31;
    const int tileN = blockIdx.x * 64;
    const int tileM = blockIdx.y * 64 + wave * 16;
    const int b     = blockIdx.z;

    const float* Bm = mla + (size_t)b * COUT_MLA * HW;

    v8f acc[4];
    #pragma unroll
    for (int nt = 0; nt < 4; ++nt)
        acc[nt] = (v8f){0.f, 0.f, 0.f, 0.f, 0.f, 0.f, 0.f, 0.f};

    gemm_tiles(wp16, COUT_MLA, Bm, HW, tileM, tileN, COUT_MLA, sB, acc);

    const int half = lane >> 4, l16 = lane & 15;
    const float* xb = x   + (size_t)b * CIN * HW;
    float*       ob = out + (size_t)b * CIN * HW;
    #pragma unroll
    for (int r = 0; r < 8; ++r) {
        const int o  = tileM + half * 8 + r;
        const float sc = gamma[o] * rsqrtf(var[o] + 1e-5f);
        const float mu = mean[o], bb = beta[o];
        #pragma unroll
        for (int nt = 0; nt < 4; ++nt) {
            const int n = tileN + nt * 16 + l16;
            const float y = (acc[nt][r] - mu) * sc + bb;
            ob[(size_t)o * HW + n] = xb[(size_t)o * HW + n] + y;
        }
    }
}

// ---------------------------------------------------------------------------
extern "C" void kernel_launch(void* const* d_in, const int* in_sizes, int n_in,
                              void* d_out, int out_size, void* d_ws, size_t ws_size,
                              hipStream_t stream)
{
    const float* x       = (const float*)d_in[0];
    const float* w_qkv   = (const float*)d_in[1];
    const float* w_dw    = (const float*)d_in[2];
    const float* w_pw    = (const float*)d_in[3];
    const float* w_proj  = (const float*)d_in[4];
    const float* bn_g    = (const float*)d_in[5];
    const float* bn_b    = (const float*)d_in[6];
    const float* bn_m    = (const float*)d_in[7];
    const float* bn_v    = (const float*)d_in[8];
    float*       out     = (float*)d_out;

    // workspace layout:
    //   qkv_ms f32 [B,1536,HW]   = 384 MB
    //   mla    f32 [B, 512,HW]   = 128 MB
    //   wq16   f16 [768,256]     = 384 KB
    //   wp16   f16 [256,512]     = 256 KB
    float*     qkv_ms = (float*)d_ws;
    float*     mla    = qkv_ms + (size_t)BATCH * CMS * HW;
    _Float16*  wq16   = (_Float16*)(mla + (size_t)BATCH * COUT_MLA * HW);
    _Float16*  wp16   = wq16 + (size_t)TD3 * CIN;

    // 0) pre-convert GEMM weights to f16
    k_cvt_f16<<<(TD3 * CIN + 255) / 256, 256, 0, stream>>>(w_qkv, wq16, TD3 * CIN);
    k_cvt_f16<<<(CIN * COUT_MLA + 255) / 256, 256, 0, stream>>>(w_proj, wp16,
                                                                CIN * COUT_MLA);
    // 1) qkv GEMM (WMMA)
    {
        dim3 grid(HW / 64, TD3 / 64, BATCH);
        k_qkv_gemm<<<grid, 128, 0, stream>>>(x, wq16, qkv_ms);
    }
    // 2) depthwise conv + group mix -> agg channels
    {
        const int total = BATCH * 96 * HW;
        k_dw_mix<<<total / 256, 256, 0, stream>>>(qkv_ms, w_dw, w_pw, qkv_ms);
    }
    // 3) lite MLA
    {
        dim3 grid(NHEAD, BATCH);
        k_lite_mla<<<grid, 256, 0, stream>>>(qkv_ms, mla);
    }
    // 4) proj GEMM (WMMA) + BN + residual
    {
        dim3 grid(HW / 64, CIN / 64, BATCH);
        k_proj_bn_res<<<grid, 128, 0, stream>>>(mla, wp16, bn_g, bn_b, bn_m, bn_v,
                                                x, out);
    }
}
